// MultiHeadAttention_44143673869037
// MI455X (gfx1250) — matrix operand order repaired and checked by reading, not compiled, not run on hardware
//
#include <hip/hip_runtime.h>
#include <hip/hip_bf16.h>
#include <math.h>

// ---------------- CDNA5 WMMA types ----------------
typedef __bf16 bf16_t;
typedef bf16_t bf16x16 __attribute__((ext_vector_type(16)));
typedef float  v8f     __attribute__((ext_vector_type(8)));

union FragBF {
    bf16x16 v;
    uint4 q[2];
    unsigned int u[8];
    unsigned short h[16];
};

__device__ __forceinline__ unsigned short f32_to_bf16_rne(float f) {
    unsigned int u = __float_as_uint(f);
    unsigned int r = 0x7FFFu + ((u >> 16) & 1u);
    u += r;
    return (unsigned short)(u >> 16);
}

__device__ __forceinline__ v8f wmma_bf16(const FragBF& a, const FragBF& b, v8f c) {
    return __builtin_amdgcn_wmma_f32_16x16x32_bf16(false, a.v, false, b.v,
                                                   (short)0, c, false, false);
}

// Problem constants
#define BN   4
#define SQ   2048
#define DM   1024
#define HH   16
#define DK   64
#define QT   16            // q rows per attention workgroup
#define SROW 2052          // LDS f32 score row stride (floats)
#define PROW 2056          // LDS bf16 prob row stride (halves), 16B-aligned rows

// LDS byte offsets inside k_attention's dynamic allocation
#define O_SC   0
#define O_PB   (QT * SROW * 4)                 // 131328
#define O_CTX  (O_PB + QT * PROW * 2)          // +65792 = 197120
#define O_RINV (O_CTX + QT * DK * 4)           // +4096  = 201216
#define O_QLDS (O_RINV + 64)                   // 201280 (16B aligned)
#define QLSTRIDE 72                            // halves per Q-tile LDS row (padded)
#define LDS_BYTES (O_QLDS + QT * QLSTRIDE * 2) // 203584

// ---------------- Kernel 1: f32 -> bf16 convert ----------------
__global__ void k_f32_to_bf16(const float* __restrict__ src,
                              unsigned short* __restrict__ dst, int n) {
    int i = blockIdx.x * blockDim.x + threadIdx.x;
    int stride = gridDim.x * blockDim.x;
    for (; i < n; i += stride) dst[i] = f32_to_bf16_rne(src[i]);
}

// ---------------- Kernel 2: NT GEMM  Y = A (MxK) * W^T (NxK) + bias -------
// Wave tile: 16x64 (one A fragment reused against 4 B fragments).
// MODE 0: bf16 out, row-major [M][N]
// MODE 1: bf16 out, V transposed per head: Vt[(b*16+h)*64+d][s]
// MODE 2: f32 out = acc + bias + resid   (output projection + residual)
template<int MODE>
__global__ void k_gemm_nt(const unsigned short* __restrict__ A,
                          const unsigned short* __restrict__ W,
                          const float* __restrict__ bias,
                          const float* __restrict__ resid,
                          void* __restrict__ out,
                          int M, int N, int K) {
    const int lane = threadIdx.x & 31;
    const int wave = threadIdx.x >> 5;
    const int nsup = N >> 6;                     // 64-wide super-tiles
    int t = blockIdx.x * 8 + wave;
    if (t >= (M >> 4) * nsup) return;            // whole-wave uniform exit
    const int mt = t / nsup, nt = t % nsup;

    const int kA = (lane >> 4) * 16;             // contiguous k order (matches B)
    const int kB = (lane >> 4) * 16;             // contiguous k order (matches A)
    const unsigned short* ap = A + (size_t)((mt << 4) + (lane & 15)) * K;
    const unsigned short* wp[4];
#pragma unroll
    for (int j = 0; j < 4; j++)
        wp[j] = W + (size_t)((nt << 6) + j * 16 + (lane & 15)) * K;

    v8f acc[4] = {{}, {}, {}, {}};
    for (int k = 0; k < K; k += 32) {
        __builtin_prefetch(ap + k + 512, 0, 0);      // global_prefetch_b8
        __builtin_prefetch(wp[0] + k + 512, 0, 0);
        FragBF a;
        a.q[0] = *(const uint4*)(ap + k + kA);
        a.q[1] = *(const uint4*)(ap + k + kA + 8);
        FragBF b[4];
#pragma unroll
        for (int j = 0; j < 4; j++) {
            b[j].q[0] = *(const uint4*)(wp[j] + k + kB);
            b[j].q[1] = *(const uint4*)(wp[j] + k + kB + 8);
        }
#pragma unroll
        for (int j = 0; j < 4; j++) acc[j] = wmma_bf16(a, b[j], acc[j]);
    }

#pragma unroll
    for (int j = 0; j < 4; j++) {
#pragma unroll
        for (int r = 0; r < 8; r++) {
            const int orow = (mt << 4) + r + ((lane >> 4) << 3);   // C layout
            const int ocol = (nt << 6) + j * 16 + (lane & 15);
            float v = acc[j][r] + bias[ocol];
            if (MODE == 0) {
                ((unsigned short*)out)[(size_t)orow * N + ocol] = f32_to_bf16_rne(v);
            } else if (MODE == 1) {
                const int b_ = orow >> 11, s = orow & 2047;
                const int h_ = ocol >> 6,  d = ocol & 63;
                ((unsigned short*)out)[(((size_t)(b_ * HH + h_) * DK + d) << 11) + s] =
                    f32_to_bf16_rne(v);
            } else {
                ((float*)out)[(size_t)orow * N + ocol] =
                    v + resid[(size_t)orow * N + ocol];
            }
        }
    }
}

// ---------------- Kernel 3: fused scores + softmax + attn-out + P@V -------
// grid = (SQ/QT, H, B), block = 256 (8 waves)
__global__ void k_attention(const unsigned short* __restrict__ Qp,
                            const unsigned short* __restrict__ Kp,
                            const unsigned short* __restrict__ Vt,
                            const unsigned char* __restrict__ mask,
                            float* __restrict__ attn_out,
                            unsigned short* __restrict__ ctx) {
    extern __shared__ char ldsraw[];
    float*          sc     = (float*)(ldsraw + O_SC);    // [QT][SROW] f32 scores/probs
    unsigned short* pb     = (unsigned short*)(ldsraw + O_PB);   // [QT][PROW] bf16 probs
    float*          ctxbuf = (float*)(ldsraw + O_CTX);   // [QT][DK]
    float*          rowinv = (float*)(ldsraw + O_RINV);  // [QT]
    unsigned short* qlds   = (unsigned short*)(ldsraw + O_QLDS); // [QT][QLSTRIDE]

    const int lane = threadIdx.x & 31;
    const int wave = threadIdx.x >> 5;
    const int qt = blockIdx.x, h = blockIdx.y, b = blockIdx.z;
    const int q0 = qt * QT;

    // ---- Async DMA the 16x64 Q tile into LDS (8 bytes per thread) ----
    {
        const int row  = threadIdx.x >> 4;
        const int coff = (threadIdx.x & 15) * 4;               // halves
        const unsigned short* gp =
            Qp + ((size_t)(b * SQ + q0 + row)) * DM + h * DK + coff;
        unsigned int ldsoff = O_QLDS + row * (QLSTRIDE * 2) + (threadIdx.x & 15) * 8;
        asm volatile("global_load_async_to_lds_b64 %0, %1, off"
                     :: "v"(ldsoff), "v"(gp) : "memory");
    }
    for (int i = threadIdx.x; i < QT * DK; i += 256) ctxbuf[i] = 0.f;
    asm volatile("s_wait_asynccnt 0" ::: "memory");
    __syncthreads();

    // ---- Phase 1: scores = (Q K^T) * 1/8, masked, into LDS ----
    {
        FragBF a0, a1;   // Q tile 16 x dk(64): two 16x32 fragments, from LDS
        const unsigned short* qp = qlds + (lane & 15) * QLSTRIDE + ((lane >> 4) * 16);
        a0.q[0] = *(const uint4*)(qp);
        a0.q[1] = *(const uint4*)(qp + 8);
        a1.q[0] = *(const uint4*)(qp + 32);
        a1.q[1] = *(const uint4*)(qp + 40);

        for (int kb = wave; kb < SQ / 16; kb += 8) {
            const unsigned short* bp = Kp
                + ((size_t)(b * SQ + kb * 16 + (lane & 15))) * DM + h * DK
                + ((lane >> 4) * 16);
            FragBF b0, b1;
            b0.q[0] = *(const uint4*)(bp);
            b0.q[1] = *(const uint4*)(bp + 8);
            b1.q[0] = *(const uint4*)(bp + 32);
            b1.q[1] = *(const uint4*)(bp + 40);
            v8f s = {};
            s = wmma_bf16(a0, b0, s);
            s = wmma_bf16(a1, b1, s);
#pragma unroll
            for (int r = 0; r < 8; r++) {
                const int rl  = r + ((lane >> 4) << 3);
                const int col = kb * 16 + (lane & 15);
                float v = s[r] * 0.125f;                       // 1/sqrt(64)
                if (mask[((size_t)b * SQ + q0 + rl) * SQ + col]) v = -__builtin_inff();
                sc[rl * SROW + col] = v;
            }
        }
    }
    __syncthreads();

    // ---- Phase 2: per-row softmax; f32 probs in sc, bf16 probs in pb ----
    for (int r = wave; r < QT; r += 8) {
        float mx = -__builtin_inff();
        for (int c = lane; c < SQ; c += 32) mx = fmaxf(mx, sc[r * SROW + c]);
        for (int o = 16; o > 0; o >>= 1) mx = fmaxf(mx, __shfl_xor(mx, o, 32));
        float sum = 0.f;
        for (int c = lane; c < SQ; c += 32) {
            float e = __expf(sc[r * SROW + c] - mx);
            sc[r * SROW + c] = e;                              // unnormalized p (f32)
            pb[r * PROW + c] = f32_to_bf16_rne(e);             // unnormalized p (bf16)
            sum += e;
        }
        for (int o = 16; o > 0; o >>= 1) sum += __shfl_xor(sum, o, 32);
        if (lane == 0) rowinv[r] = (sum > 0.f) ? 1.f / sum : 0.f;
    }
    __syncthreads();

    // ---- Phase 3: stream normalized attention (f32) to HBM, single pass ----
    {
        float* ao = attn_out + (((size_t)(h * BN + b)) * SQ + q0) * SQ;
        for (int idx = threadIdx.x; idx < QT * SQ; idx += 256) {
            const int r = idx >> 11, c = idx & 2047;
            ao[(size_t)r * SQ + c] = sc[r * SROW + c] * rowinv[r];
        }
    }

    // ---- Phase 4: ctx = P @ V; A frags straight from bf16 LDS (ds_load_b128)
    {
        const int nb = wave & 3;             // dk tile (4 x 16)
        const int kh = wave >> 2;            // K half
        const int kbA = (lane >> 4) * 16;
        const unsigned short* prow = pb + (lane & 15) * PROW;
        const unsigned short* vp = Vt
            + (((size_t)(b * HH + h) * DK) + nb * 16 + (lane & 15)) * SQ
            + ((lane >> 4) * 16);
        v8f acc = {};
        for (int k = kh * (SQ / 2); k < (kh + 1) * (SQ / 2); k += 32) {
            FragBF a, bv;
            a.q[0] = *(const uint4*)(prow + k + kbA);
            a.q[1] = *(const uint4*)(prow + k + kbA + 8);
            bv.q[0] = *(const uint4*)(vp + k);
            bv.q[1] = *(const uint4*)(vp + k + 8);
            acc = wmma_bf16(a, bv, acc);
        }
#pragma unroll
        for (int r = 0; r < 8; r++) {
            const int rl  = r + ((lane >> 4) << 3);
            const int col = nb * 16 + (lane & 15);
            atomicAdd(&ctxbuf[rl * DK + col], acc[r]);         // ds_add_f32
        }
    }
    __syncthreads();

    for (int idx = threadIdx.x; idx < QT * DK; idx += 256) {
        const int r = idx >> 6, c = idx & 63;
        float v = ctxbuf[idx] * rowinv[r];
        ctx[((size_t)(b * SQ + q0 + r)) * DM + h * DK + c] = f32_to_bf16_rne(v);
    }
}

// ---------------- Kernel 4: LayerNorm over D=1024 ----------------
__global__ void k_layernorm(const float* __restrict__ Y,
                            const float* __restrict__ gamma,
                            const float* __restrict__ beta,
                            float* __restrict__ out) {
    __shared__ float sred[16];
    const int m = blockIdx.x;
    const float* y = Y + (size_t)m * DM;
    float s = 0.f, s2 = 0.f;
    for (int c = threadIdx.x; c < DM; c += 256) {
        float v = y[c]; s += v; s2 += v * v;
    }
    for (int o = 16; o > 0; o >>= 1) { s += __shfl_xor(s, o, 32); s2 += __shfl_xor(s2, o, 32); }
    const int wave = threadIdx.x >> 5, lane = threadIdx.x & 31;
    if (lane == 0) { sred[wave] = s; sred[8 + wave] = s2; }
    __syncthreads();
    if (threadIdx.x == 0) {
        float a = 0.f, b = 0.f;
        for (int i = 0; i < 8; i++) { a += sred[i]; b += sred[8 + i]; }
        sred[0] = a; sred[8] = b;
    }
    __syncthreads();
    const float mean = sred[0] * (1.f / DM);
    const float var  = sred[8] * (1.f / DM) - mean * mean;
    const float rinv = rsqrtf(var + 1e-5f);
    for (int c = threadIdx.x; c < DM; c += 256)
        out[(size_t)m * DM + c] = gamma[c] * (y[c] - mean) * rinv + beta[c];
}

// ---------------- Host-side launch ----------------
extern "C" void kernel_launch(void* const* d_in, const int* in_sizes, int n_in,
                              void* d_out, int out_size, void* d_ws, size_t ws_size,
                              hipStream_t stream) {
    (void)in_sizes; (void)n_in; (void)out_size; (void)ws_size;
    const float* query = (const float*)d_in[0];
    const float* keyi  = (const float*)d_in[1];
    const float* value = (const float*)d_in[2];
    const unsigned char* mask = (const unsigned char*)d_in[3];
    const float* Wq = (const float*)d_in[4];
    const float* bq = (const float*)d_in[5];
    const float* Wk = (const float*)d_in[6];
    const float* bk = (const float*)d_in[7];
    const float* Wv = (const float*)d_in[8];
    const float* bv = (const float*)d_in[9];
    const float* Wo = (const float*)d_in[10];
    const float* bo = (const float*)d_in[11];
    const float* gamma = (const float*)d_in[12];
    const float* beta  = (const float*)d_in[13];

    float* out  = (float*)d_out;
    float* attn = out + (size_t)BN * SQ * DM;     // outputs concatenated flat

    char* ws = (char*)d_ws;
    const size_t MiB = 1ull << 20;
    unsigned short* Xq  = (unsigned short*)(ws + 0 * MiB);
    unsigned short* Xk  = (unsigned short*)(ws + 16 * MiB);
    unsigned short* Xv  = (unsigned short*)(ws + 32 * MiB);
    unsigned short* Wqb = (unsigned short*)(ws + 48 * MiB);
    unsigned short* Wkb = (unsigned short*)(ws + 50 * MiB);
    unsigned short* Wvb = (unsigned short*)(ws + 52 * MiB);
    unsigned short* Wob = (unsigned short*)(ws + 54 * MiB);
    unsigned short* Qp  = (unsigned short*)(ws + 56 * MiB);
    unsigned short* Kp  = (unsigned short*)(ws + 72 * MiB);
    unsigned short* Vt  = (unsigned short*)(ws + 88 * MiB);
    unsigned short* ctx = (unsigned short*)(ws + 104 * MiB);
    float*          Yb  = (float*)(ws + 120 * MiB);

    const int NX = BN * SQ * DM;   // 8.4M
    const int NW = DM * DM;        // 1M
    k_f32_to_bf16<<<2048, 256, 0, stream>>>(query, Xq, NX);
    k_f32_to_bf16<<<2048, 256, 0, stream>>>(keyi,  Xk, NX);
    k_f32_to_bf16<<<2048, 256, 0, stream>>>(value, Xv, NX);
    k_f32_to_bf16<<<512, 256, 0, stream>>>(Wq, Wqb, NW);
    k_f32_to_bf16<<<512, 256, 0, stream>>>(Wk, Wkb, NW);
    k_f32_to_bf16<<<512, 256, 0, stream>>>(Wv, Wvb, NW);
    k_f32_to_bf16<<<512, 256, 0, stream>>>(Wo, Wob, NW);

    const int M = BN * SQ;
    const int gemm_blocks = (M / 16) * (DM / 64) / 8;   // one 16x64 tile per wave
    k_gemm_nt<0><<<gemm_blocks, 256, 0, stream>>>(Xq, Wqb, bq, nullptr, Qp, M, DM, DM);
    k_gemm_nt<0><<<gemm_blocks, 256, 0, stream>>>(Xk, Wkb, bk, nullptr, Kp, M, DM, DM);
    k_gemm_nt<1><<<gemm_blocks, 256, 0, stream>>>(Xv, Wvb, bv, nullptr, Vt, M, DM, DM);

    dim3 agrid(SQ / QT, HH, BN);
    k_attention<<<agrid, 256, LDS_BYTES, stream>>>(Qp, Kp, Vt, mask, attn, ctx);

    k_gemm_nt<2><<<gemm_blocks, 256, 0, stream>>>(ctx, Wob, bo, query, Yb, M, DM, DM);
    k_layernorm<<<M, 256, 0, stream>>>(Yb, gamma, beta, out);
}
